// SelfAttention_13864154432024
// MI455X (gfx1250) — matrix operand order repaired and checked by reading, not compiled, not run on hardware
//
#include <hip/hip_runtime.h>
#include <math.h>

// ---- problem sizes ----
#define NB 4
#define SS 2048
#define EE 1024
#define HH 16
#define DD 64

typedef __attribute__((ext_vector_type(16))) __bf16  v16bf;
typedef __attribute__((ext_vector_type(8)))  __bf16  v8bf;
typedef __attribute__((ext_vector_type(8)))  float   v8f;

__device__ __forceinline__ v8f wmma_bf16(v16bf a, v16bf b, v8f c) {
    return __builtin_amdgcn_wmma_f32_16x16x32_bf16(
        /*neg_a=*/false, a, /*neg_b=*/false, b,
        /*c_mod=*/(short)0, c, /*reuse_a=*/false, /*reuse_b=*/false);
}

__device__ __forceinline__ v16bf cat8(v8bf lo, v8bf hi) {
    v16bf a;
#pragma unroll
    for (int i = 0; i < 8; ++i) { a[i] = lo[i]; a[8 + i] = hi[i]; }
    return a;
}

// pack A fragment from fp32: elements [0..7] from p[0..7], [8..15] from p[8..15]
__device__ __forceinline__ v16bf cvtA(const float* p) {
    v16bf a;
#pragma unroll
    for (int i = 0; i < 8; ++i) {
        a[i]     = (__bf16)p[i];        // native v_cvt_pk_bf16_f32 (RNE)
        a[8 + i] = (__bf16)p[i + 8];
    }
    return a;
}

// -------------------------------------------------------------------------
// Weight conversion fp32 -> bf16.  wqkv layout: [Wq | Wk | Wv] (4096 each)
// -------------------------------------------------------------------------
__global__ __launch_bounds__(256) void cvtw_kernel(
    const float* __restrict__ wq, const float* __restrict__ wk,
    const float* __restrict__ wv, const float* __restrict__ wo,
    __bf16* __restrict__ wqkv, __bf16* __restrict__ wob)
{
    int i = blockIdx.x * 256 + threadIdx.x;
    if (i < 3 * DD * DD) {
        const float* s = (i < DD * DD) ? wq : (i < 2 * DD * DD ? wk : wv);
        wqkv[i] = (__bf16)s[i & (DD * DD - 1)];
    } else {
        int j = i - 3 * DD * DD;   // 0 .. EE*EE-1
        wob[j] = (__bf16)wo[j];
    }
}

// -------------------------------------------------------------------------
// Q/K/V projection: rows = 16 heads of one token, cols = e (64), K = d (64).
// dst layout: vtrans==0 -> [n,h,s,e];  vtrans==1 -> [n,h,e,s] (pre-transposed
// so the attention PV B-operand is a contiguous 32B load per lane).
// Both layouts have stride SS*DD between heads, so stores are branch-free:
// dst[base + v*SS*DD] with one uniform select for the base.
// 8 waves: (w&1) = token within pair, (w>>1) = e-tile.
// -------------------------------------------------------------------------
__global__ __launch_bounds__(256) void proj_kernel(
    const float* __restrict__ src, const __bf16* __restrict__ wbf,
    __bf16* __restrict__ dst, int vtrans)
{
    const int tid = threadIdx.x;
    const int w = tid >> 5, lane = tid & 31;
    const int lh = lane >> 4, ln = lane & 15;
    const long nl = (long)blockIdx.x * 2 + (w & 1);
    const int n = (int)(nl >> 11);          // / SS
    const int l = (int)(nl & (SS - 1));
    const int ebase = (w >> 1) * 16;
    const float* x = src + nl * EE;         // token row; + head*64 + d

    v8f acc = {};
#pragma unroll
    for (int kb = 0; kb < DD; kb += 32) {
        // A: row = head = ln; per-lane K chunk lh*16 .. lh*16+15 (fp32 -> bf16)
        v16bf a = cvtA(x + ln * DD + kb + lh * 16);
        // B[d,e] = W[e,d]: col e = ebase+ln, 16 consecutive d per lane
        v16bf b = *reinterpret_cast<const v16bf*>(
            wbf + (ebase + ln) * DD + kb + lh * 16);
        acc = wmma_bf16(a, b, acc);
    }
    const int e = ebase + ln;
    const int m0 = lh * 8;                  // first output head of this lane
    // head stride is SS*DD in BOTH layouts -> branch-free store loop
    const size_t base = vtrans
        ? (((size_t)n * HH + m0) * DD + e) * SS + l      // [n,h,e,s]
        : (((size_t)n * HH + m0) * SS + l) * DD + e;     // [n,h,s,e]
    __bf16* dp = dst + base;
#pragma unroll
    for (int v = 0; v < 8; ++v)
        dp[(size_t)v * SS * DD] = (__bf16)acc[v];
}

// -------------------------------------------------------------------------
// Attention for one (n,h) x 16-query stripe.  Full score row kept in LDS.
// -------------------------------------------------------------------------
#define SCP 2056   // fp32 score row pitch (32B-aligned chunks, bank-spread)

__global__ __launch_bounds__(256) void attn_kernel(
    const __bf16* __restrict__ qp, const __bf16* __restrict__ kp,
    const __bf16* __restrict__ vpt, const int* __restrict__ mask,
    __bf16* __restrict__ ao)
{
    __shared__ float sc[16 * SCP];            // 131584 B
    __shared__ float obuf[16 * DD];           // 4096 B
    __shared__ float red[16 * 16];
    __shared__ float rsum[16];                // reciprocal row sums

    const int tid = threadIdx.x;
    const int w = tid >> 5, lane = tid & 31;
    const int lh = lane >> 4, ln = lane & 15;
    const int qbase = blockIdx.x * 16;
    const int nh = blockIdx.y;                // n*16 + h
    const int n = nh >> 4, h = nh & 15;
    const __bf16* Q = qp + ((size_t)nh * SS + qbase) * DD;
    const __bf16* K = kp + (size_t)nh * SS * DD;
    const float scale = 0.03125f;             // 1/sqrt(E), E=1024

    // ---- Phase 1: S = Q K^T, mask+scale folded into the LDS store ----
    v16bf aq[2];
#pragma unroll
    for (int kb = 0; kb < 2; ++kb) {
        const __bf16* qr = Q + ln * DD + kb * 32 + lh * 16;
        aq[kb] = cat8(*reinterpret_cast<const v8bf*>(qr),
                      *reinterpret_cast<const v8bf*>(qr + 8));
    }
    for (int jt = w * 16; jt < SS; jt += 128) {
        v8f acc = {};
#pragma unroll
        for (int kb = 0; kb < 2; ++kb) {
            // B[d,j] = K[j,d]: col j = jt+ln, 16 consecutive d per lane
            v16bf b = *reinterpret_cast<const v16bf*>(
                K + (size_t)(jt + ln) * DD + kb * 32 + lh * 16);
            acc = wmma_bf16(aq[kb], b, acc);
        }
        // column mask (key jt+ln) is uniform across the 8 rows of this lane
        const int mv = mask[n * SS + jt + ln];
        const float ms = mv ? scale : 0.0f;
        const float mb = mv ? 0.0f : -1.0e30f;
#pragma unroll
        for (int v = 0; v < 8; ++v)
            sc[(v + lh * 8) * SCP + jt + ln] = fmaf(acc[v], ms, mb);
    }
    __syncthreads();

    // ---- Phase 2: softmax statistics (fp32) ----
    {
        const int r = tid >> 4, c = tid & 15;
        float lmax = -3.0e38f;
        for (int j = c; j < SS; j += 16)           // read-only max scan
            lmax = fmaxf(lmax, sc[r * SCP + j]);
        red[r * 16 + c] = lmax;
        __syncthreads();
        float rmax = -3.0e38f;
#pragma unroll
        for (int i = 0; i < 16; ++i) rmax = fmaxf(rmax, red[r * 16 + i]);
        __syncthreads();
        float lsum = 0.f;
        for (int j = c; j < SS; j += 16) {
            float e = __expf(sc[r * SCP + j] - rmax);
            sc[r * SCP + j] = e;                   // unnormalized P
            lsum += e;
        }
        red[r * 16 + c] = lsum;
        __syncthreads();
        if (c == 0) {
            float t = 0.f;
#pragma unroll
            for (int i = 0; i < 16; ++i) t += red[r * 16 + i];
            rsum[r] = 1.0f / t;
        }
        __syncthreads();
    }

    // ---- Phase 3: O = P V.  wave -> (dchunk = w&3, jsub = w>>2). ----
    // V is pre-transposed [n,h,d,s]: B operand = contiguous 32B global load.
    const int dchunk = w & 3, jsub = w >> 2;
    const __bf16* Vt = vpt + ((size_t)nh * DD + dchunk * 16 + ln) * SS;
    v8f oacc = {};
    const int j0 = jsub * (SS / 2);
    for (int jb = j0; jb < j0 + SS / 2; jb += 32) {
        // A: P rows (row = ln), K chunk lh*16 .. lh*16+15 within 32-key slice
        v16bf a = cvtA(&sc[ln * SCP + jb + lh * 16]);
        // B[j,d]: col d = dchunk*16+ln (folded into Vt), 16 consecutive j
        v16bf b = *reinterpret_cast<const v16bf*>(Vt + jb + lh * 16);
        oacc = wmma_bf16(a, b, oacc);
    }
    if (jsub == 1) {
#pragma unroll
        for (int v = 0; v < 8; ++v)
            obuf[(v + lh * 8) * DD + dchunk * 16 + ln] = oacc[v];
    }
    __syncthreads();
    if (jsub == 0) {
#pragma unroll
        for (int v = 0; v < 8; ++v) {
            int m = v + lh * 8;
            float o = (oacc[v] + obuf[m * DD + dchunk * 16 + ln]) * rsum[m];
            // ao layout [n, s, h, d]  (== [n,s,e] with e = h*64+d)
            size_t idx = (((size_t)n * SS + qbase + m) * HH + h) * DD
                       + dchunk * 16 + ln;
            ao[idx] = (__bf16)o;
        }
    }
}

// -------------------------------------------------------------------------
// Output projection: out[r,o] = sum_e ao[r,e]*Wo[o,e] + bo[o], fp32 out.
// WG = 16 rows x 128 cols (8 waves, one 16x16 tile each, K=1024).
// -------------------------------------------------------------------------
__global__ __launch_bounds__(256) void outproj_kernel(
    const __bf16* __restrict__ ao, const __bf16* __restrict__ wob,
    const float* __restrict__ bo, float* __restrict__ out)
{
    const int tid = threadIdx.x;
    const int w = tid >> 5, lane = tid & 31;
    const int lh = lane >> 4, ln = lane & 15;
    const long rbase = (long)blockIdx.x * 16;
    const int obase = blockIdx.y * 128 + w * 16;
    const __bf16* arow = ao + (size_t)(rbase + ln) * EE;
    const __bf16* brow = wob + (size_t)(obase + ln) * EE;

    v8f acc = {};
#pragma unroll 4
    for (int kb = 0; kb < EE; kb += 32) {
        const __bf16* ap = arow + kb + lh * 16;
        v16bf a = cat8(*reinterpret_cast<const v8bf*>(ap),
                       *reinterpret_cast<const v8bf*>(ap + 8));
        v16bf b = *reinterpret_cast<const v16bf*>(brow + kb + lh * 16);
        acc = wmma_bf16(a, b, acc);
    }
    const float bv = bo[obase + ln];
#pragma unroll
    for (int v = 0; v < 8; ++v)
        out[(rbase + v + lh * 8) * EE + obase + ln] = acc[v] + bv;
}

// -------------------------------------------------------------------------
extern "C" void kernel_launch(void* const* d_in, const int* in_sizes, int n_in,
                              void* d_out, int out_size, void* d_ws, size_t ws_size,
                              hipStream_t stream) {
    const float* values = (const float*)d_in[0];
    const float* keysi  = (const float*)d_in[1];
    const float* query  = (const float*)d_in[2];
    const int*   mask   = (const int*)d_in[3];
    const float* Wv     = (const float*)d_in[4];
    const float* Wk     = (const float*)d_in[5];
    const float* Wq     = (const float*)d_in[6];
    const float* Wo     = (const float*)d_in[7];
    const float* bo     = (const float*)d_in[8];
    float* out = (float*)d_out;

    // workspace layout (bf16)
    const size_t PER = (size_t)NB * HH * SS * DD;     // 8.39M elements
    __bf16* qp   = (__bf16*)d_ws;
    __bf16* kp   = qp + PER;
    __bf16* vpt  = kp + PER;                          // [n,h,d,s]
    __bf16* ao   = vpt + PER;                         // [n,s,e]
    __bf16* wob  = ao + (size_t)NB * SS * EE;         // Wo bf16
    __bf16* wqkv = wob + (size_t)EE * EE;             // Wq|Wk|Wv bf16

    cvtw_kernel<<<(3 * DD * DD + EE * EE) / 256, 256, 0, stream>>>(
        Wq, Wk, Wv, Wo, wqkv, wob);

    proj_kernel<<<NB * SS / 2, 256, 0, stream>>>(query,  wqkv,               qp,  0);
    proj_kernel<<<NB * SS / 2, 256, 0, stream>>>(keysi,  wqkv + DD * DD,     kp,  0);
    proj_kernel<<<NB * SS / 2, 256, 0, stream>>>(values, wqkv + 2 * DD * DD, vpt, 1);

    attn_kernel<<<dim3(SS / 16, NB * HH), 256, 0, stream>>>(qp, kp, vpt, mask, ao);

    outproj_kernel<<<dim3(NB * SS / 16, EE / 128), 256, 0, stream>>>(
        ao, wob, bo, out);
}
